// SwiGLUFFN_59665685676692
// MI455X (gfx1250) — 8-bit matrix operand order repaired and checked by reading, compiled, not run on hardware
//
#include <hip/hip_runtime.h>
#include <math.h>

#define DIM 2048
#define HID 5461
#define HP  5504
#define TOK 8192

typedef __attribute__((ext_vector_type(8))) int v8i;


__device__ __forceinline__ float wave_max_f(float m) {
#pragma unroll
  for (int off = 16; off; off >>= 1) m = fmaxf(m, __shfl_xor(m, off, 32));
  return m;
}
__device__ __forceinline__ int wave_max_i(int m) {
#pragma unroll
  for (int off = 16; off; off >>= 1) m = max(m, __shfl_xor(m, off, 32));
  return m;
}
__device__ __forceinline__ float adc_q(int raw, float s) {
  return fminf(fmaxf(rintf((float)raw / s), -512.0f), 511.0f) * s;
}
__device__ __forceinline__ signed char q_w8(float v, float inv_ws) {
  return (signed char)(int)fminf(fmaxf(rintf(v * inv_ws), -128.0f), 127.0f);
}

__global__ void k_init(unsigned int* scal) {
  if (threadIdx.x < 8) scal[threadIdx.x] = 0u;
}

__global__ void k_absmax(const float* __restrict__ in, long long n,
                         unsigned int* __restrict__ out_bits) {
  long long stride = (long long)gridDim.x * blockDim.x;
  float m = 0.0f;
  for (long long i = (long long)blockIdx.x * blockDim.x + threadIdx.x; i < n; i += stride)
    m = fmaxf(m, fabsf(in[i]));
  m = wave_max_f(m);
  if ((threadIdx.x & 31) == 0) atomicMax(out_bits, __float_as_uint(m));
}

__global__ void k_quant_w12(const float* __restrict__ w, const unsigned int* __restrict__ scal,
                            signed char* __restrict__ w1q, signed char* __restrict__ w2q) {
  int k = blockIdx.x * blockDim.x + threadIdx.x;
  int n = blockIdx.y;
  float ws = fmaxf(__uint_as_float(scal[1]) / 127.0f, 1e-8f);
  float inv = 1.0f / ws;
  signed char q1 = 0, q2 = 0;
  if (n < HID) {
    q1 = q_w8(w[(size_t)n * DIM + k], inv);
    q2 = q_w8(w[(size_t)(n + HID) * DIM + k], inv);
  }
  size_t o = (size_t)n * DIM + k;
  w1q[o] = q1;
  w2q[o] = q2;
}

__global__ void k_quant_w3(const float* __restrict__ w, const unsigned int* __restrict__ scal,
                           signed char* __restrict__ w3q) {
  int k = blockIdx.x * blockDim.x + threadIdx.x;
  int r = blockIdx.y;
  if (k >= HP) return;
  float ws = fmaxf(__uint_as_float(scal[2]) / 127.0f, 1e-8f);
  float inv = 1.0f / ws;
  signed char q = 0;
  if (k < HID) q = q_w8(w[(size_t)r * HID + k], inv);
  w3q[(size_t)r * HP + k] = q;
}

__global__ void k_quant_x(const float* __restrict__ x, const unsigned int* __restrict__ scal,
                          signed char* __restrict__ xm, unsigned char* __restrict__ xl) {
  float as = fmaxf(__uint_as_float(scal[0]) / 2047.0f, 1e-8f);
  float inv = 1.0f / as;
  long long total = (long long)TOK * DIM;
  long long stride = (long long)gridDim.x * blockDim.x;
  for (long long i = (long long)blockIdx.x * blockDim.x + threadIdx.x; i < total; i += stride) {
    float q = fminf(fmaxf(rintf(x[i] * inv), -2048.0f), 2047.0f);
    float mb = fminf(fmaxf(floorf(q * 0.0625f), -128.0f), 127.0f);
    xm[i] = (signed char)(int)mb;
    xl[i] = (unsigned char)(int)(q - mb * 16.0f);
  }
}

__global__ void k_quant_h(const float* __restrict__ hid, const unsigned int* __restrict__ scal,
                          signed char* __restrict__ hm, unsigned char* __restrict__ hl) {
  int c = blockIdx.x * blockDim.x + threadIdx.x;
  int r = blockIdx.y;
  if (c >= HP) return;
  float as = fmaxf(__uint_as_float(scal[3]) / 2047.0f, 1e-8f);
  float inv = 1.0f / as;
  signed char m8 = 0;
  unsigned char l8 = 0;
  size_t i = (size_t)r * HP + c;
  if (c < HID) {
    float q = fminf(fmaxf(rintf(hid[i] * inv), -2048.0f), 2047.0f);
    float mb = fminf(fmaxf(floorf(q * 0.0625f), -128.0f), 127.0f);
    m8 = (signed char)(int)mb;
    l8 = (unsigned char)(int)(q - mb * 16.0f);
  }
  hm[i] = m8;
  hl[i] = l8;
}

__device__ __forceinline__ v8i load_a_frag(const int2* __restrict__ s, int m, int h) {
  v8i a;
#pragma unroll
  for (int p = 0; p < 4; ++p) {
    int2 t = s[m * 8 + (p >> 1) * 4 + h * 2 + (p & 1)];
    a[2 * p] = t.x;
    a[2 * p + 1] = t.y;
  }
  return a;
}
__device__ __forceinline__ v8i load_b_frag(const signed char* __restrict__ p) {
  int4 u = *(const int4*)p;
  int4 v = *(const int4*)(p + 32);
  v8i b;
  b[0] = u.x; b[1] = u.y; b[2] = u.z; b[3] = u.w;
  b[4] = v.x; b[5] = v.y; b[6] = v.z; b[7] = v.w;
  return b;
}

#define WMMA_IU8 __builtin_amdgcn_wmma_i32_16x16x64_iu8

template <int PASS>
__global__ void __launch_bounds__(256)
k_gemm12(const signed char* __restrict__ Am, const unsigned char* __restrict__ Al,
         const signed char* __restrict__ B1, const signed char* __restrict__ B2,
         const float* __restrict__ bias, unsigned int* __restrict__ scal,
         float* __restrict__ hid) {
  __shared__ int4 sA[2][8][64];
  const int lane = threadIdx.x & 31;
  const int wave = threadIdx.x >> 5;
  const int h = lane >> 4;
  const int ln = lane & 15;
  const int mBase = blockIdx.y * 64;
  const int nBase = (blockIdx.x * 8 + wave) * 16;

  v8i c1m[4] = {}, c1l[4] = {}, c2m[4] = {}, c2l[4] = {};

  const int lmt = (threadIdx.x >> 6) & 3;
  const int lsl = (threadIdx.x >> 5) & 1;
  const int lidx = threadIdx.x & 31;
  const int ltile = lsl * 4 + lmt;
  const signed char* gA = lsl ? (const signed char*)Al : Am;
  const size_t gAoff = (size_t)(mBase + lmt * 16 + (lidx >> 1)) * DIM + (lidx & 1) * 32;

  const signed char* pB1 = B1 + (size_t)(nBase + ln) * DIM + h * 16;
  const signed char* pB2 = B2 + (size_t)(nBase + ln) * DIM + h * 16;

  {
    int4 r0 = *(const int4*)(gA + gAoff);
    int4 r1 = *(const int4*)(gA + gAoff + 16);
    sA[0][ltile][lidx * 2] = r0;
    sA[0][ltile][lidx * 2 + 1] = r1;
  }
  __syncthreads();

  int buf = 0;
  for (int k0 = 0; k0 < DIM; k0 += 64) {
    if (k0 + 64 < DIM) {
      int4 r0 = *(const int4*)(gA + gAoff + k0 + 64);
      int4 r1 = *(const int4*)(gA + gAoff + k0 + 80);
      sA[buf ^ 1][ltile][lidx * 2] = r0;
      sA[buf ^ 1][ltile][lidx * 2 + 1] = r1;
      __builtin_prefetch(pB1 + k0 + 64, 0, 1);
      __builtin_prefetch(pB2 + k0 + 64, 0, 1);
    }

    v8i b1 = load_b_frag(pB1 + k0);
    v8i b2 = load_b_frag(pB2 + k0);
#pragma unroll
    for (int mt = 0; mt < 4; ++mt) {
      v8i am = load_a_frag((const int2*)&sA[buf][mt][0], ln, h);
      v8i al = load_a_frag((const int2*)&sA[buf][4 + mt][0], ln, h);
      c1m[mt] = WMMA_IU8(true,  am, true, b1, c1m[mt], false, false);
      c1l[mt] = WMMA_IU8(false, al, true, b1, c1l[mt], false, false);
      c2m[mt] = WMMA_IU8(true,  am, true, b2, c2m[mt], false, false);
      c2l[mt] = WMMA_IU8(false, al, true, b2, c2l[mt], false, false);
    }
    __syncthreads();
    buf ^= 1;
  }

  if (PASS == 0) {
    int mM = 0, mL = 0;
#pragma unroll
    for (int mt = 0; mt < 4; ++mt)
#pragma unroll
      for (int r = 0; r < 8; ++r) {
        mM = max(mM, max(abs(c1m[mt][r]), abs(c2m[mt][r])));
        mL = max(mL, max(abs(c1l[mt][r]), abs(c2l[mt][r])));
      }
    mM = wave_max_i(mM);
    mL = wave_max_i(mL);
    if (lane == 0) {
      atomicMax((int*)&scal[4], mM);
      atomicMax((int*)&scal[5], mL);
    }
  } else {
    float aS = fmaxf(__uint_as_float(scal[0]) / 2047.0f, 1e-8f);
    float wS = fmaxf(__uint_as_float(scal[1]) / 127.0f, 1e-8f);
    float sM = (float)(int)scal[4] / 511.0f + 1e-8f;
    float sL = (float)(int)scal[5] / 511.0f + 1e-8f;
    float gs = aS * wS * 16.0f;
    float gl = aS * wS;
    int n = nBase + ln;
    float hmax = 0.0f;
    if (n < HID) {
      float b1v = bias[n], b2v = bias[HID + n];
#pragma unroll
      for (int mt = 0; mt < 4; ++mt)
#pragma unroll
        for (int r = 0; r < 8; ++r) {
          int m = mBase + mt * 16 + r + 8 * h;
          float y1 = adc_q(c1m[mt][r], sM) * gs + adc_q(c1l[mt][r], sL) * gl + b1v;
          float y2 = adc_q(c2m[mt][r], sM) * gs + adc_q(c2l[mt][r], sL) * gl + b2v;
          float hv = (y1 / (1.0f + __expf(-y1))) * y2;
          hid[(size_t)m * HP + n] = hv;
          hmax = fmaxf(hmax, fabsf(hv));
        }
    }
    hmax = wave_max_f(hmax);
    if (lane == 0) atomicMax(&scal[3], __float_as_uint(hmax));
  }
}

template <int PASS>
__global__ void __launch_bounds__(256)
k_gemm3(const signed char* __restrict__ Am, const unsigned char* __restrict__ Al,
        const signed char* __restrict__ B, const float* __restrict__ bias,
        unsigned int* __restrict__ scal, float* __restrict__ out) {
  __shared__ int4 sA[2][8][64];
  const int lane = threadIdx.x & 31;
  const int wave = threadIdx.x >> 5;
  const int h = lane >> 4;
  const int ln = lane & 15;
  const int mBase = blockIdx.y * 64;
  const int nBase = (blockIdx.x * 8 + wave) * 16;

  v8i cM[4] = {}, cL[4] = {};

  const int lmt = (threadIdx.x >> 6) & 3;
  const int lsl = (threadIdx.x >> 5) & 1;
  const int lidx = threadIdx.x & 31;
  const int ltile = lsl * 4 + lmt;
  const signed char* gA = lsl ? (const signed char*)Al : Am;
  const size_t gAoff = (size_t)(mBase + lmt * 16 + (lidx >> 1)) * HP + (lidx & 1) * 32;

  const signed char* pB = B + (size_t)(nBase + ln) * HP + h * 16;

  {
    int4 r0 = *(const int4*)(gA + gAoff);
    int4 r1 = *(const int4*)(gA + gAoff + 16);
    sA[0][ltile][lidx * 2] = r0;
    sA[0][ltile][lidx * 2 + 1] = r1;
  }
  __syncthreads();

  int buf = 0;
  for (int k0 = 0; k0 < HP; k0 += 64) {
    if (k0 + 64 < HP) {
      int4 r0 = *(const int4*)(gA + gAoff + k0 + 64);
      int4 r1 = *(const int4*)(gA + gAoff + k0 + 80);
      sA[buf ^ 1][ltile][lidx * 2] = r0;
      sA[buf ^ 1][ltile][lidx * 2 + 1] = r1;
      __builtin_prefetch(pB + k0 + 64, 0, 1);
    }

    v8i bf = load_b_frag(pB + k0);
#pragma unroll
    for (int mt = 0; mt < 4; ++mt) {
      v8i am = load_a_frag((const int2*)&sA[buf][mt][0], ln, h);
      v8i al = load_a_frag((const int2*)&sA[buf][4 + mt][0], ln, h);
      cM[mt] = WMMA_IU8(true,  am, true, bf, cM[mt], false, false);
      cL[mt] = WMMA_IU8(false, al, true, bf, cL[mt], false, false);
    }
    __syncthreads();
    buf ^= 1;
  }

  if (PASS == 0) {
    int mM = 0, mL = 0;
#pragma unroll
    for (int mt = 0; mt < 4; ++mt)
#pragma unroll
      for (int r = 0; r < 8; ++r) {
        mM = max(mM, abs(cM[mt][r]));
        mL = max(mL, abs(cL[mt][r]));
      }
    mM = wave_max_i(mM);
    mL = wave_max_i(mL);
    if (lane == 0) {
      atomicMax((int*)&scal[6], mM);
      atomicMax((int*)&scal[7], mL);
    }
  } else {
    float aS = fmaxf(__uint_as_float(scal[3]) / 2047.0f, 1e-8f);
    float wS = fmaxf(__uint_as_float(scal[2]) / 127.0f, 1e-8f);
    float sM = (float)(int)scal[6] / 511.0f + 1e-8f;
    float sL = (float)(int)scal[7] / 511.0f + 1e-8f;
    float gs = aS * wS * 16.0f;
    float gl = aS * wS;
    int n = nBase + ln;
    float bv = bias[n];
#pragma unroll
    for (int mt = 0; mt < 4; ++mt)
#pragma unroll
      for (int r = 0; r < 8; ++r) {
        int m = mBase + mt * 16 + r + 8 * h;
        out[(size_t)m * DIM + n] = adc_q(cM[mt][r], sM) * gs + adc_q(cL[mt][r], sL) * gl + bv;
      }
  }
}

extern "C" void kernel_launch(void* const* d_in, const int* in_sizes, int n_in,
                              void* d_out, int out_size, void* d_ws, size_t ws_size,
                              hipStream_t stream) {
  const float* x    = (const float*)d_in[0];
  const float* w12w = (const float*)d_in[1];
  const float* w12b = (const float*)d_in[2];
  const float* w3w  = (const float*)d_in[3];
  const float* w3b  = (const float*)d_in[4];
  float* out = (float*)d_out;

  char* ws = (char*)d_ws;
  size_t o = 0;
  auto alloc = [&](size_t sz) { size_t r = o; o += (sz + 255) & ~(size_t)255; return r; };

  unsigned int* scal = (unsigned int*)(ws + alloc(64 * sizeof(unsigned int)));
  signed char*   xm  = (signed char*)(ws + alloc((size_t)TOK * DIM));
  unsigned char* xl  = (unsigned char*)(ws + alloc((size_t)TOK * DIM));
  signed char*   w1q = (signed char*)(ws + alloc((size_t)HP * DIM));
  signed char*   w2q = (signed char*)(ws + alloc((size_t)HP * DIM));
  signed char*   w3q = (signed char*)(ws + alloc((size_t)DIM * HP));
  float*         hid = (float*)(ws + alloc((size_t)TOK * HP * sizeof(float)));
  signed char*   hm  = (signed char*)(ws + alloc((size_t)TOK * HP));
  unsigned char* hl  = (unsigned char*)(ws + alloc((size_t)TOK * HP));
  (void)in_sizes; (void)n_in; (void)out_size; (void)ws_size;

  k_init<<<1, 32, 0, stream>>>(scal);

  k_absmax<<<1024, 256, 0, stream>>>(x,    (long long)TOK * DIM,      scal + 0);
  k_absmax<<<1024, 256, 0, stream>>>(w12w, (long long)2 * HID * DIM,  scal + 1);
  k_absmax<<<512,  256, 0, stream>>>(w3w,  (long long)DIM * HID,      scal + 2);

  k_quant_w12<<<dim3(DIM / 256, HP), 256, 0, stream>>>(w12w, scal, w1q, w2q);
  k_quant_w3 <<<dim3((HP + 255) / 256, DIM), 256, 0, stream>>>(w3w, scal, w3q);
  k_quant_x  <<<2048, 256, 0, stream>>>(x, scal, xm, xl);

  dim3 g12(HP / 128, TOK / 64);
  k_gemm12<0><<<g12, 256, 0, stream>>>(xm, xl, w1q, w2q, w12b, scal, hid);
  k_gemm12<1><<<g12, 256, 0, stream>>>(xm, xl, w1q, w2q, w12b, scal, hid);

  k_quant_h<<<dim3((HP + 255) / 256, TOK), 256, 0, stream>>>(hid, scal, hm, hl);

  dim3 g3(DIM / 128, TOK / 64);
  k_gemm3<0><<<g3, 256, 0, stream>>>(hm, hl, w3q, w3b, scal, out);
  k_gemm3<1><<<g3, 256, 0, stream>>>(hm, hl, w3q, w3b, scal, out);
}
